// GAT_51342039056534
// MI455X (gfx1250) — matrix operand order repaired and checked by reading, not compiled, not run on hardware
//
#include <hip/hip_runtime.h>
#include <hip/hip_bf16.h>

// ---------------------------------------------------------------------------
// GAT (3 GATConv layers + edge MLP) for MI455X / gfx1250.
//  - Dense GEMMs (h = x @ W^T) on the WMMA pipe: bf16 in, f32 accum,
//    V_WMMA_F32_16X16X32_BF16. Register-tiled 16x64 per wave (NT=4): one A
//    fragment is reused against 4 B fragments -> 10 b128 loads per 4 WMMAs
//    instead of 16, and A is re-read 4x less from L2.
//  - Attention softmax / segment reductions in fp32 via hardware atomics
//    (atomic f32 add; atomic u32 max with order-preserving float keys).
//  - Edge aggregation is byte-dominant; h1 (102MB) fits the 192MB L2, so the
//    random gathers/atomic-adds stay on-die.
// ---------------------------------------------------------------------------

typedef __attribute__((ext_vector_type(16))) __bf16 v16bf;
typedef __attribute__((ext_vector_type(8)))  __bf16 v8bf;
typedef __attribute__((ext_vector_type(8)))  float  v8f;

#define NNODES 50000
#define NEDGES 400000
#define ETOT   (NEDGES + NNODES)   // edges + self loops

// ---- order-preserving float <-> uint key for atomic segment max ------------
__device__ __forceinline__ unsigned fkey(float f) {
  unsigned u = __float_as_uint(f);
  return (u & 0x80000000u) ? ~u : (u | 0x80000000u);
}
__device__ __forceinline__ float funkey(unsigned k) {
  return (k & 0x80000000u) ? __uint_as_float(k & 0x7FFFFFFFu)
                           : __uint_as_float(~k);
}

// ---------------------------------------------------------------------------
// elementwise fp32 -> bf16
__global__ void k_cvt_bf16(const float* __restrict__ in,
                           __bf16* __restrict__ out, long long n) {
  long long i = (long long)blockIdx.x * blockDim.x + threadIdx.x;
  long long st = (long long)gridDim.x * blockDim.x;
  for (; i < n; i += st) out[i] = (__bf16)in[i];
}

// W3 [8,128] -> bf16 [16,128] zero-padded (pad N to one WMMA tile)
__global__ void k_cvt_w3(const float* __restrict__ w3, __bf16* __restrict__ out) {
  int i = blockIdx.x * blockDim.x + threadIdx.x;
  if (i >= 16 * 128) return;
  out[i] = ((i >> 7) < 8) ? (__bf16)w3[i] : (__bf16)0.0f;
}

__global__ void k_zero(unsigned* __restrict__ p, long long n) {
  long long i = (long long)blockIdx.x * blockDim.x + threadIdx.x;
  long long st = (long long)gridDim.x * blockDim.x;
  for (; i < n; i += st) p[i] = 0u;
}

// ---------------------------------------------------------------------------
// C[M,Nf] (f32) = A[M,K] (bf16, row-major) x W^T, W = [Nf,K] (bf16 row-major).
// One 16x(16*NT) supertile per wave; 8 waves / block; K stepped by 32.
// A frag: lane L -> row L&15, K = (L>>4)*16 + 0..15 (matches B)
// B frag: lane L -> col L&15, K = (L>>4)*16 + 0..15 == contiguous in W row.
// C frag: VGPR j -> row j + 8*(L>>4), col L&15.
template <int NT>
__global__ __launch_bounds__(256)
void k_wmma_gemm(const __bf16* __restrict__ A, const __bf16* __restrict__ W,
                 float* __restrict__ C, int M, int Nf, int K) {
  int lane = threadIdx.x & 31;
  int wave = threadIdx.x >> 5;
  int ntSuper = Nf / (16 * NT);
  int tiles = (M >> 4) * ntSuper;
  int tile = blockIdx.x * 8 + wave;          // uniform per wave -> EXEC all-1s
  if (tile >= tiles) return;
  int tm = tile / ntSuper, tn = tile - tm * ntSuper;
  int r = lane & 15, hh = lane >> 4;

  const __bf16* ap = A + (size_t)(tm * 16 + r) * K + (hh << 4);
  const __bf16* bp = W + (size_t)(tn * 16 * NT + r) * K + (hh << 4);

  v8f acc[NT];
#pragma unroll
  for (int t = 0; t < NT; ++t) acc[t] = (v8f){};

  for (int kb = 0; kb < K; kb += 32) {
    v8bf alo = *(const v8bf*)(ap + kb);        // K = base+0..7   (16B)
    v8bf ahi = *(const v8bf*)(ap + kb + 8);    // K = base+8..15  (16B)
    v16bf a;
#pragma unroll
    for (int i = 0; i < 8; ++i) { a[i] = alo[i]; a[i + 8] = ahi[i]; }
#pragma unroll
    for (int t = 0; t < NT; ++t) {             // reuse A across 4 B tiles
      v16bf b = *(const v16bf*)(bp + (size_t)t * 16 * K + kb);
      acc[t] = __builtin_amdgcn_wmma_f32_16x16x32_bf16(
          false, a, false, b, (short)0, acc[t], false, false);
    }
  }
#pragma unroll
  for (int t = 0; t < NT; ++t) {
    float* cp = C + (size_t)(tm * 16 + hh * 8) * Nf + tn * 16 * NT + t * 16 + r;
#pragma unroll
    for (int j = 0; j < 8; ++j) cp[(size_t)j * Nf] = acc[t][j];
  }
}

// ---------------------------------------------------------------------------
// per-(node,head) attention coefficients  s = <h, a_s>, d = <h, a_d>
__global__ void k_node_alpha(const float* __restrict__ h, int ldh,
                             const float* __restrict__ as_,
                             const float* __restrict__ ad_,
                             float* __restrict__ s, float* __restrict__ d,
                             int H, int C) {
  int idx = blockIdx.x * blockDim.x + threadIdx.x;
  if (idx >= NNODES * H) return;
  int n = idx / H, head = idx - n * H;
  const float* hp = h + (size_t)n * ldh + head * C;
  const float* ap = as_ + head * C;
  const float* bp = ad_ + head * C;
  float ss = 0.f, dd = 0.f;
  for (int c = 0; c < C; ++c) { float v = hp[c]; ss += v * ap[c]; dd += v * bp[c]; }
  s[idx] = ss; d[idx] = dd;
}

// edge pass 1: leaky-relu logits + segment max over dst (uint-keyed atomicMax)
__global__ void k_edge_logits(const int* __restrict__ ei,
                              const float* __restrict__ s,
                              const float* __restrict__ d,
                              float* __restrict__ ev,
                              unsigned* __restrict__ mkey, int H) {
  long long total = (long long)ETOT * H;
  long long idx = (long long)blockIdx.x * blockDim.x + threadIdx.x;
  long long st = (long long)gridDim.x * blockDim.x;
  for (; idx < total; idx += st) {
    int e = (int)(idx / H), head = (int)(idx - (long long)e * H);
    int src = (e < NEDGES) ? ei[e] : (e - NEDGES);
    int dst = (e < NEDGES) ? ei[NEDGES + e] : (e - NEDGES);
    float v = s[src * H + head] + d[dst * H + head];
    v = (v > 0.f) ? v : 0.2f * v;                 // leaky_relu(0.2)
    ev[idx] = v;
    atomicMax(&mkey[dst * H + head], fkey(v));
  }
}

// edge pass 2: ex = exp(e - max[dst]); segment sum of ex over dst
__global__ void k_edge_expsum(const int* __restrict__ ei,
                              const float* __restrict__ ev,
                              const unsigned* __restrict__ mkey,
                              float* __restrict__ exv,
                              float* __restrict__ den, int H) {
  long long total = (long long)ETOT * H;
  long long idx = (long long)blockIdx.x * blockDim.x + threadIdx.x;
  long long st = (long long)gridDim.x * blockDim.x;
  for (; idx < total; idx += st) {
    int e = (int)(idx / H), head = (int)(idx - (long long)e * H);
    int dst = (e < NEDGES) ? ei[NEDGES + e] : (e - NEDGES);
    float x = __expf(ev[idx] - funkey(mkey[dst * H + head]));
    exv[idx] = x;
    atomicAdd(&den[dst * H + head], x);
  }
}

// edge pass 3: out[dst] += (ex/den[dst]) * h[src]   (F-wide, coalesced over f)
__global__ void k_edge_agg(const int* __restrict__ ei,
                           const float* __restrict__ exv,
                           const float* __restrict__ den,
                           const float* __restrict__ h, int ldh,
                           float* __restrict__ out, int ldo, int H, int C) {
  int F = H * C;
  long long total = (long long)ETOT * F;
  long long idx = (long long)blockIdx.x * blockDim.x + threadIdx.x;
  long long st = (long long)gridDim.x * blockDim.x;
  for (; idx < total; idx += st) {
    long long e = idx / F;
    int f = (int)(idx - e * F);
    int head = f / C;
    int ie = (int)e;
    int src = (ie < NEDGES) ? ei[ie] : (ie - NEDGES);
    int dst = (ie < NEDGES) ? ei[NEDGES + ie] : (ie - NEDGES);
    float w = exv[e * H + head] / den[(size_t)dst * H + head];
    atomicAdd(&out[(size_t)dst * ldo + f], w * h[(size_t)src * ldh + f]);
  }
}

// bias + ELU; optionally emit f32 and/or bf16 (next-layer WMMA operand)
__global__ void k_bias_act(const float* __restrict__ acc, int ldacc,
                           const float* __restrict__ bias, int F,
                           float* __restrict__ outF, __bf16* __restrict__ outB,
                           int ldout) {
  long long total = (long long)NNODES * F;
  long long idx = (long long)blockIdx.x * blockDim.x + threadIdx.x;
  long long st = (long long)gridDim.x * blockDim.x;
  for (; idx < total; idx += st) {
    long long n = idx / F;
    int f = (int)(idx - n * F);
    float v = acc[n * ldacc + f] + bias[f];
    v = (v > 0.f) ? v : (__expf(v) - 1.f);        // ELU(alpha=1)
    if (outF) outF[n * ldout + f] = v;
    if (outB) outB[n * ldout + f] = (__bf16)v;
  }
}

// final per-edge MLP: z[19] = [h3[src], h3[dst], ea, yr, qt]; 19->16 relu ->1
__global__ __launch_bounds__(256)
void k_edge_mlp(const int* __restrict__ ei, const float* __restrict__ h3,
                const float* __restrict__ ea, const float* __restrict__ yr,
                const float* __restrict__ qt, const float* __restrict__ f1w,
                const float* __restrict__ f1b, const float* __restrict__ f2w,
                const float* __restrict__ f2b, float* __restrict__ out) {
  __shared__ float sw[337];                 // 16*19 + 16 + 16 + 1
  for (int i = threadIdx.x; i < 337; i += blockDim.x) {
    float v;
    if (i < 304) v = f1w[i];
    else if (i < 320) v = f1b[i - 304];
    else if (i < 336) v = f2w[i - 320];
    else v = f2b[0];
    sw[i] = v;
  }
  __syncthreads();
  int e = blockIdx.x * blockDim.x + threadIdx.x;
  if (e >= NEDGES) return;
  int s = ei[e], d = ei[NEDGES + e];
  float z[19];
#pragma unroll
  for (int i = 0; i < 8; ++i) z[i] = h3[(size_t)s * 8 + i];
#pragma unroll
  for (int i = 0; i < 8; ++i) z[8 + i] = h3[(size_t)d * 8 + i];
  z[16] = ea[e]; z[17] = yr[e]; z[18] = qt[e];
  float r = sw[336];
#pragma unroll
  for (int j = 0; j < 16; ++j) {
    float o = sw[304 + j];
#pragma unroll
    for (int k = 0; k < 19; ++k) o += sw[j * 19 + k] * z[k];
    o = (o > 0.f) ? o : 0.f;                       // relu
    r += sw[320 + j] * o;
  }
  out[e] = r;
}

// ---------------------------------------------------------------------------
static unsigned gblk(long long total, int tpb) {
  long long b = (total + tpb - 1) / tpb;
  return (unsigned)(b > 0x3FFFFFFFLL ? 0x3FFFFFFFLL : b);
}

static void run_layer(hipStream_t stream, const __bf16* Ain, int K,
                      const __bf16* Wb, int Nf, int ntile, const int* ei,
                      const float* as_, const float* ad_, const float* bias,
                      int H, int C, int ldh, int ldo, float* hbuf, float* obuf,
                      float* sbuf, float* dbuf, unsigned* mkey, float* den,
                      float* ev, float* exv, float* actF, __bf16* actB,
                      int ldact) {
  const int T = 256;
  int F = H * C;
  int tiles = (NNODES / 16) * (Nf / (16 * ntile));
  if (ntile == 4)
    k_wmma_gemm<4><<<(tiles + 7) / 8, 256, 0, stream>>>(Ain, Wb, hbuf, NNODES, Nf, K);
  else
    k_wmma_gemm<1><<<(tiles + 7) / 8, 256, 0, stream>>>(Ain, Wb, hbuf, NNODES, Nf, K);
  k_node_alpha<<<gblk((long long)NNODES * H, T), T, 0, stream>>>(
      hbuf, ldh, as_, ad_, sbuf, dbuf, H, C);
  k_zero<<<gblk((long long)NNODES * H, T), T, 0, stream>>>(mkey, (long long)NNODES * H);
  k_zero<<<gblk((long long)NNODES * H, T), T, 0, stream>>>((unsigned*)den, (long long)NNODES * H);
  k_zero<<<gblk((long long)NNODES * F, T), T, 0, stream>>>((unsigned*)obuf, (long long)NNODES * F);
  long long eh = (long long)ETOT * H;
  k_edge_logits<<<gblk(eh, T), T, 0, stream>>>(ei, sbuf, dbuf, ev, mkey, H);
  k_edge_expsum<<<gblk(eh, T), T, 0, stream>>>(ei, ev, mkey, exv, den, H);
  k_edge_agg<<<gblk((long long)ETOT * F, T), T, 0, stream>>>(
      ei, exv, den, hbuf, ldh, obuf, ldo, H, C);
  k_bias_act<<<gblk((long long)NNODES * F, T), T, 0, stream>>>(
      obuf, ldo, bias, F, actF, actB, ldact);
}

extern "C" void kernel_launch(void* const* d_in, const int* in_sizes, int n_in,
                              void* d_out, int out_size, void* d_ws,
                              size_t ws_size, hipStream_t stream) {
  const float* x   = (const float*)d_in[0];
  const int*   ei  = (const int*)d_in[1];
  const float* ea  = (const float*)d_in[2];
  const float* yr  = (const float*)d_in[3];
  const float* qt  = (const float*)d_in[4];
  const float* W1  = (const float*)d_in[5];
  const float* a1s = (const float*)d_in[6];
  const float* a1d = (const float*)d_in[7];
  const float* b1  = (const float*)d_in[8];
  const float* W2  = (const float*)d_in[9];
  const float* a2s = (const float*)d_in[10];
  const float* a2d = (const float*)d_in[11];
  const float* b2  = (const float*)d_in[12];
  const float* W3  = (const float*)d_in[13];
  const float* a3s = (const float*)d_in[14];
  const float* a3d = (const float*)d_in[15];
  const float* b3  = (const float*)d_in[16];
  const float* f1w = (const float*)d_in[17];
  const float* f1b = (const float*)d_in[18];
  const float* f2w = (const float*)d_in[19];
  const float* f2b = (const float*)d_in[20];
  float* out = (float*)d_out;

  // ---- workspace layout (256B aligned regions) -----------------------------
  char* ws = (char*)d_ws;
  size_t off = 0;
  auto take = [&](size_t bytes) -> void* {
    void* p = ws + off;
    off = (off + bytes + 255) & ~(size_t)255;
    return p;
  };
  __bf16* W1b  = (__bf16*)take((size_t)512 * 128 * 2);
  __bf16* W2b  = (__bf16*)take((size_t)128 * 512 * 2);
  __bf16* W3b  = (__bf16*)take((size_t)16 * 128 * 2);
  __bf16* Abf  = (__bf16*)take((size_t)NNODES * 512 * 2);  // layer input (bf16)
  float*  hbuf = (float*)take((size_t)NNODES * 512 * 4);   // GEMM out (pre-attn)
  float*  obuf = (float*)take((size_t)NNODES * 512 * 4);   // aggregation accum
  float*  sbuf = (float*)take((size_t)NNODES * 4 * 4);
  float*  dbuf = (float*)take((size_t)NNODES * 4 * 4);
  unsigned* mkey = (unsigned*)take((size_t)NNODES * 4 * 4);
  float*  den  = (float*)take((size_t)NNODES * 4 * 4);
  float*  ev   = (float*)take((size_t)ETOT * 4 * 4);
  float*  exv  = (float*)take((size_t)ETOT * 4 * 4);
  float*  h3a  = (float*)take((size_t)NNODES * 8 * 4);
  (void)in_sizes; (void)n_in; (void)out_size; (void)ws_size;

  const int T = 256;
  // stage weights + input activations as bf16 WMMA operands
  k_cvt_bf16<<<gblk(512 * 128, T), T, 0, stream>>>(W1, W1b, 512 * 128);
  k_cvt_bf16<<<gblk(128 * 512, T), T, 0, stream>>>(W2, W2b, 128 * 512);
  k_cvt_w3<<<gblk(16 * 128, T), T, 0, stream>>>(W3, W3b);
  k_cvt_bf16<<<gblk((long long)NNODES * 128, T), T, 0, stream>>>(
      x, Abf, (long long)NNODES * 128);

  // layer 1: 128 -> 4x128 (F=512), NT=4 supertiles
  run_layer(stream, Abf, 128, W1b, 512, 4, ei, a1s, a1d, b1, 4, 128, 512, 512,
            hbuf, obuf, sbuf, dbuf, mkey, den, ev, exv, nullptr, Abf, 512);
  // layer 2: 512 -> 4x32 (F=128), NT=4 supertiles
  run_layer(stream, Abf, 512, W2b, 128, 4, ei, a2s, a2d, b2, 4, 32, 128, 128,
            hbuf, obuf, sbuf, dbuf, mkey, den, ev, exv, nullptr, Abf, 128);
  // layer 3: 128 -> 1x8 (GEMM N padded to 16; ld of GEMM out = 16, agg ld = 8)
  run_layer(stream, Abf, 128, W3b, 16, 1, ei, a3s, a3d, b3, 1, 8, 16, 8,
            hbuf, obuf, sbuf, dbuf, mkey, den, ev, exv, h3a, nullptr, 8);

  // final edge scorer
  k_edge_mlp<<<gblk(NEDGES, 256), 256, 0, stream>>>(
      ei, h3a, ea, yr, qt, f1w, f1b, f2w, f2b, out);
}
